// MHA_rotary_37864431682657
// MI455X (gfx1250) — matrix operand order repaired and checked by reading, not compiled, not run on hardware
//
#include <hip/hip_runtime.h>

// ---------------------------------------------------------------------------
// MHA with rotary embedding for MI455X (gfx1250).
// bf16 WMMA for all matmuls, TDM async-tensor staging for attention K tiles.
// B=2, T=2048, C=A=1024, H=16, D=64, ROT=32.
// ---------------------------------------------------------------------------

typedef unsigned short u16b;
typedef __bf16 bf16_t;
typedef bf16_t v16bf __attribute__((ext_vector_type(16)));
typedef float  v8f   __attribute__((ext_vector_type(8)));

typedef unsigned int ui32x4 __attribute__((ext_vector_type(4)));
typedef int          i32x8  __attribute__((ext_vector_type(8)));
typedef int          i32x4  __attribute__((ext_vector_type(4)));

#define B_  2
#define T_  2048
#define H_  16
#define D_  64
#define C_  1024
#define M_  (B_ * T_)   // 4096 rows for the projection GEMMs

static __device__ __forceinline__ u16b f2bf(float f) {
  union { float f; unsigned u; } v; v.f = f;
  unsigned u = v.u;
  return (u16b)((u + 0x7FFFu + ((u >> 16) & 1u)) >> 16);   // RNE
}
static __device__ __forceinline__ float bf2f(u16b h) {
  union { unsigned u; float f; } v; v.u = ((unsigned)h) << 16;
  return v.f;
}

union FragAB { v16bf v; uint4 q[2]; u16b h[16]; };

static __device__ __forceinline__ v8f wmma_bf16(const FragAB& a, const FragAB& b, v8f c) {
  return __builtin_amdgcn_wmma_f32_16x16x32_bf16(
      false, a.v, false, b.v, (short)0, c, false, false);
}

// ---------------------------------------------------------------------------
// TDM: async DMA of a 2D bf16 tile (tile_dim0=64 elems wide, 32 rows,
// row stride 64 elems) from global memory into LDS at lds_off.
// Descriptor layout per CDNA5 ISA (D# groups 0/1).
// Toolchain uses the 6-arg builtin: (g0, g1, g2, g3, gX, cpol).
// ---------------------------------------------------------------------------
static __device__ __forceinline__ void tdm_load_k_tile(const u16b* gptr, unsigned lds_off) {
  const unsigned long long ga = (unsigned long long)(uintptr_t)gptr;
  ui32x4 g0;
  g0[0] = 1u;                                              // count=1, user mode
  g0[1] = lds_off;                                         // lds_addr (bytes)
  g0[2] = (unsigned)(ga & 0xFFFFFFFFu);                    // global_addr[31:0]
  g0[3] = (unsigned)((ga >> 32) & 0x01FFFFFFu)             // global_addr[56:32]
        | (2u << 30);                                      // type = 2 ("image")
  i32x8 g1;
  g1[0] = 0x00010000;            // workgroup_mask=0, data_size=1 (2 bytes)
  g1[1] = (int)(64u << 16);      // atomic_barrier_addr=0 ; tensor_dim0[15:0]=64
  g1[2] = (int)(2048u << 16);    // tensor_dim0[31:16]=0  ; tensor_dim1[15:0]=2048
  g1[3] = (int)(64u << 16);      // tensor_dim1[31:16]=0  ; tile_dim0=64
  g1[4] = 32;                    // tile_dim1=32 ; tile_dim2=0
  g1[5] = 64;                    // tensor_dim0_stride[31:0]=64
  g1[6] = 0;                     // stride hi / tensor_dim1_stride lo
  g1[7] = 0;
  const i32x4 z4 = {0, 0, 0, 0};
  const i32x8 z8 = {0, 0, 0, 0, 0, 0, 0, 0};
  __builtin_amdgcn_tensor_load_to_lds(g0, g1, z4, z4, z8, 0);
}

// ---------------------------------------------------------------------------
// f32 -> bf16 conversion
// ---------------------------------------------------------------------------
__global__ void cvt_kernel(const float* __restrict__ in, u16b* __restrict__ out, int n) {
  int i = blockIdx.x * blockDim.x + threadIdx.x;
  const int stride = gridDim.x * blockDim.x;
  for (; i < n; i += stride) out[i] = f2bf(in[i]);
}

// ---------------------------------------------------------------------------
// GEMM: out = A(MxK bf16) @ W(KxN bf16) + bias.
// MODE 0: f32 row-major output (final projection).
// MODE 1: bf16 output scattered to (B,H,T,D) layout (QKV projections).
// Block: 256 threads = 8 waves, tile 128x128; wave tile 32x64.
// ---------------------------------------------------------------------------
template <int MODE>
__global__ __launch_bounds__(256)
void gemm_bf16(const u16b* __restrict__ A,
               const u16b* __restrict__ W,
               const float* __restrict__ bias,
               float* __restrict__ outF,
               u16b* __restrict__ outH)
{
  const int N = 1024, K = 1024;
  __shared__ __align__(16) u16b As[128][32];   // [m][k]
  __shared__ __align__(16) u16b Bs[128][32];   // [n][k] (transposed)

  const int tid  = threadIdx.x;
  const int lane = tid & 31;
  const int wv   = tid >> 5;        // wave 0..7
  const int wm   = wv & 3;          // 4 waves along M
  const int wn   = wv >> 2;         // 2 waves along N
  const int half = (lane >> 4) & 1;
  const int l16  = lane & 15;

  const int m0 = blockIdx.y * 128;
  const int n0 = blockIdx.x * 128;

  v8f acc[2][4];
  const v8f vzero = {0.f,0.f,0.f,0.f,0.f,0.f,0.f,0.f};
  for (int i = 0; i < 2; ++i)
    for (int j = 0; j < 4; ++j) acc[i][j] = vzero;

  const int arow = tid >> 1;          // 0..127
  const int ak   = (tid & 1) * 16;    // 0 / 16
  const int bk   = tid >> 3;          // 0..31
  const int bn   = (tid & 7) * 16;    // 0..112

  for (int k0 = 0; k0 < K; k0 += 32) {
    // stage A tile (row-major)
    {
      const uint4* src = (const uint4*)(A + (size_t)(m0 + arow) * K + k0 + ak);
      uint4* dst = (uint4*)(&As[arow][ak]);
      dst[0] = src[0];
      dst[1] = src[1];
    }
    // stage B tile transposed: Bs[n][k]
    {
      union { uint4 q[2]; u16b h[16]; } tmp;
      const uint4* src = (const uint4*)(W + (size_t)(k0 + bk) * N + n0 + bn);
      tmp.q[0] = src[0];
      tmp.q[1] = src[1];
#pragma unroll
      for (int e = 0; e < 16; ++e) Bs[bn + e][bk] = tmp.h[e];
    }
    // warm L2/L0 for next K-step (global_prefetch_b8)
    if (k0 + 32 < K) {
      __builtin_prefetch(A + (size_t)(m0 + arow) * K + (k0 + 32) + ak, 0, 3);
      __builtin_prefetch(W + (size_t)(k0 + 32 + bk) * N + n0 + bn, 0, 3);
    }
    __syncthreads();

    FragAB af[2], bf[4];
#pragma unroll
    for (int sm = 0; sm < 2; ++sm) {
      const int row = wm * 32 + sm * 16 + l16;
      af[sm].q[0] = *(const uint4*)(&As[row][half * 8]);        // K = half*8 .. +7
      af[sm].q[1] = *(const uint4*)(&As[row][16 + half * 8]);   // K = 16+half*8 .. +7
    }
#pragma unroll
    for (int sn = 0; sn < 4; ++sn) {
      const int col = wn * 64 + sn * 16 + l16;
      bf[sn].q[0] = *(const uint4*)(&Bs[col][half * 8]);        // K = half*8 .. +7
      bf[sn].q[1] = *(const uint4*)(&Bs[col][16 + half * 8]);   // K = 16+half*8 .. +7
    }
#pragma unroll
    for (int sm = 0; sm < 2; ++sm)
#pragma unroll
      for (int sn = 0; sn < 4; ++sn)
        acc[sm][sn] = wmma_bf16(af[sm], bf[sn], acc[sm][sn]);
    __syncthreads();
  }

  // epilogue: bias + store
#pragma unroll
  for (int sm = 0; sm < 2; ++sm) {
#pragma unroll
    for (int sn = 0; sn < 4; ++sn) {
      const int n = n0 + wn * 64 + sn * 16 + l16;
      const float bb = bias[n];
#pragma unroll
      for (int r = 0; r < 8; ++r) {
        const int m = m0 + wm * 32 + sm * 16 + r + half * 8;
        const float val = acc[sm][sn][r] + bb;
        if (MODE == 0) {
          outF[(size_t)m * N + n] = val;
        } else {
          const int b = m >> 11;       // / T_
          const int t = m & (T_ - 1);
          const int h = n >> 6;        // / D_
          const int d = n & (D_ - 1);
          outH[(((size_t)(b * H_ + h)) * T_ + t) * D_ + d] = f2bf(val);
        }
      }
    }
  }
}

// ---------------------------------------------------------------------------
// Rotary embedding, in-place on Q and K (B,H,T,D) bf16. ROT=32, BASE=2048=2^11.
// ---------------------------------------------------------------------------
__global__ __launch_bounds__(256)
void rotary_kernel(u16b* __restrict__ Q, u16b* __restrict__ Kk)
{
  const int idx = blockIdx.x * blockDim.x + threadIdx.x;   // (b,h,t) flat
  if (idx >= B_ * H_ * T_) return;
  const int t = idx & (T_ - 1);
  u16b* qp = Q  + (size_t)idx * D_;
  u16b* kp = Kk + (size_t)idx * D_;
#pragma unroll
  for (int i = 0; i < 16; ++i) {
    const float invf = exp2f(-11.0f * (float)i / 16.0f);   // 2048^(-i/16)
    const float ang = (float)t * invf;
    const float c = cosf(ang);
    const float s = sinf(ang);
    {
      float a = bf2f(qp[i]), b = bf2f(qp[i + 16]);
      qp[i]      = f2bf(a * c - b * s);
      qp[i + 16] = f2bf(b * c + a * s);
    }
    {
      float a = bf2f(kp[i]), b = bf2f(kp[i + 16]);
      kp[i]      = f2bf(a * c - b * s);
      kp[i + 16] = f2bf(b * c + a * s);
    }
  }
}

// ---------------------------------------------------------------------------
// Flash attention. Grid (T/128, H, B), 256 threads = 8 waves.
// K tiles staged by the Tensor Data Mover (double-buffered, wave 0 driver);
// V staged transposed by vector loads; P relayout via per-wave LDS tiles.
// Output written bf16 into (B, T, H*D) layout for the final GEMM.
// ---------------------------------------------------------------------------
__global__ __launch_bounds__(256)
void attn_kernel(const u16b* __restrict__ Q,
                 const u16b* __restrict__ Kk,
                 const u16b* __restrict__ V,
                 u16b* __restrict__ O)
{
  __shared__ __align__(16) u16b Ksb[2][32][64];   // [buf][key][d], TDM-filled
  __shared__ __align__(16) u16b Vs[64][32];       // [d][key] (transposed)
  __shared__ __align__(16) u16b Ps[8][16][32];    // per-wave P tiles

  const int tid  = threadIdx.x;
  const int lane = tid & 31;
  const int wv   = tid >> 5;
  const int half = (lane >> 4) & 1;
  const int l16  = lane & 15;

  const int qbase = blockIdx.x * 128;
  const int h = blockIdx.y;
  const int b = blockIdx.z;
  const size_t headOff = ((size_t)(b * H_ + h)) * T_ * D_;
  const u16b* kHead = Kk + headOff;
  const unsigned ksOff = (unsigned)(uintptr_t)(&Ksb[0][0][0]);  // LDS byte offset

  // Q fragments (16 rows x 64 dims, as two K=32 A-fragments)
  FragAB qf[2];
  {
    const u16b* qp = Q + headOff + (size_t)(qbase + wv * 16 + l16) * D_;
#pragma unroll
    for (int c = 0; c < 2; ++c) {
      qf[c].q[0] = *(const uint4*)(qp + c * 32 + half * 8);
      qf[c].q[1] = *(const uint4*)(qp + c * 32 + 16 + half * 8);
    }
  }

  const v8f vzero = {0.f,0.f,0.f,0.f,0.f,0.f,0.f,0.f};
  v8f o[4];
#pragma unroll
  for (int i = 0; i < 4; ++i) o[i] = vzero;
  float mrow[8], lrow[8];
#pragma unroll
  for (int r = 0; r < 8; ++r) { mrow[r] = -1.0e30f; lrow[r] = 0.0f; }

  const int kvkey = tid >> 3;         // 0..31
  const int kvd   = (tid & 7) * 8;    // 0..56
  const float scale = 0.125f;         // 1/sqrt(64)
  const int kend = qbase + 128;       // causal horizon for this block

  // prime the TDM pipeline with the first K tile
  if (wv == 0) tdm_load_k_tile(kHead, ksOff);

  for (int kb = 0; kb < kend; kb += 32) {
    const int ib = (kb >> 5) & 1;

    // wave 0: issue next K tile, then wait for the current one
    if (wv == 0) {
      if (kb + 32 < kend) {
        tdm_load_k_tile(kHead + (size_t)(kb + 32) * D_,
                        ksOff + (unsigned)((((kb >> 5) + 1) & 1) * 4096u));
        __builtin_amdgcn_s_wait_tensorcnt(1);
      } else {
        __builtin_amdgcn_s_wait_tensorcnt(0);
      }
    }

    // all threads: stage V tile transposed
    {
      union { uint4 q; u16b h[8]; } tv;
      tv.q = *(const uint4*)(V + headOff + (size_t)(kb + kvkey) * D_ + kvd);
#pragma unroll
      for (int e = 0; e < 8; ++e) Vs[kvd + e][kvkey] = tv.h[e];
      if (kb + 32 < kend)
        __builtin_prefetch(V + headOff + (size_t)(kb + 32 + kvkey) * D_ + kvd, 0, 3);
    }
    __syncthreads();

    // S = Q K^T  (two 16x16 column tiles, D split into two K=32 chunks)
    v8f s[2];
#pragma unroll
    for (int nt = 0; nt < 2; ++nt) {
      FragAB kf0, kf1;
      const u16b* krow = &Ksb[ib][nt * 16 + l16][0];
      {
        kf0.q[0] = *(const uint4*)(krow + half * 8);
        kf0.q[1] = *(const uint4*)(krow + 16 + half * 8);
        kf1.q[0] = *(const uint4*)(krow + 32 + half * 8);
        kf1.q[1] = *(const uint4*)(krow + 48 + half * 8);
      }
      v8f z = vzero;
      z = wmma_bf16(qf[0], kf0, z);
      z = wmma_bf16(qf[1], kf1, z);
      s[nt] = z;
    }

    // streaming softmax update (stats are lane-uniform per 16-lane half)
#pragma unroll
    for (int r = 0; r < 8; ++r) {
      const int rowg = qbase + wv * 16 + r + half * 8;
      float s0 = ((kb + l16)      <= rowg) ? s[0][r] * scale : -1.0e30f;
      float s1 = ((kb + 16 + l16) <= rowg) ? s[1][r] * scale : -1.0e30f;
      float mx = fmaxf(s0, s1);
      mx = fmaxf(mx, __shfl_xor(mx, 1, 32));
      mx = fmaxf(mx, __shfl_xor(mx, 2, 32));
      mx = fmaxf(mx, __shfl_xor(mx, 4, 32));
      mx = fmaxf(mx, __shfl_xor(mx, 8, 32));
      const float mnew  = fmaxf(mrow[r], mx);
      const float alpha = __expf(mrow[r] - mnew);
      const float e0 = __expf(s0 - mnew);
      const float e1 = __expf(s1 - mnew);
      float sum = e0 + e1;
      sum += __shfl_xor(sum, 1, 32);
      sum += __shfl_xor(sum, 2, 32);
      sum += __shfl_xor(sum, 4, 32);
      sum += __shfl_xor(sum, 8, 32);
      lrow[r] = lrow[r] * alpha + sum;
      mrow[r] = mnew;
#pragma unroll
      for (int nt4 = 0; nt4 < 4; ++nt4) o[nt4][r] *= alpha;
      Ps[wv][r + half * 8][l16]      = f2bf(e0);
      Ps[wv][r + half * 8][16 + l16] = f2bf(e1);
    }
    __syncthreads();

    // reload P in A-fragment layout, O += P V
    FragAB pf;
    {
      const u16b* pp = &Ps[wv][l16][0];
      pf.q[0] = *(const uint4*)(pp + half * 8);
      pf.q[1] = *(const uint4*)(pp + 16 + half * 8);
    }
#pragma unroll
    for (int nt4 = 0; nt4 < 4; ++nt4) {
      FragAB vf;
      vf.q[0] = *(const uint4*)(&Vs[nt4 * 16 + l16][half * 8]);
      vf.q[1] = *(const uint4*)(&Vs[nt4 * 16 + l16][16 + half * 8]);
      o[nt4] = wmma_bf16(pf, vf, o[nt4]);
    }
    __syncthreads();
  }

  // normalize and store (B, T, H*D) bf16
#pragma unroll
  for (int r = 0; r < 8; ++r) {
    const float inv = 1.0f / lrow[r];
    const int trow = qbase + wv * 16 + r + half * 8;
    u16b* op = O + ((size_t)b * T_ + trow) * (H_ * D_) + h * D_;
#pragma unroll
    for (int nt4 = 0; nt4 < 4; ++nt4)
      op[nt4 * 16 + l16] = f2bf(o[nt4][r] * inv);
  }
}

// ---------------------------------------------------------------------------
// Launch
// ---------------------------------------------------------------------------
extern "C" void kernel_launch(void* const* d_in, const int* in_sizes, int n_in,
                              void* d_out, int out_size, void* d_ws, size_t ws_size,
                              hipStream_t stream)
{
  const float* x  = (const float*)d_in[0];
  const float* Wq = (const float*)d_in[1];
  const float* bq = (const float*)d_in[2];
  const float* Wk = (const float*)d_in[3];
  const float* bk = (const float*)d_in[4];
  const float* Wv = (const float*)d_in[5];
  const float* bv = (const float*)d_in[6];
  const float* Wo = (const float*)d_in[7];
  const float* bo = (const float*)d_in[8];
  float* out = (float*)d_out;

  const size_t NX = (size_t)M_ * C_;   // 4,194,304
  const size_t NW = (size_t)C_ * C_;   // 1,048,576

  u16b* xb  = (u16b*)d_ws;
  u16b* Wqb = xb  + NX;
  u16b* Wkb = Wqb + NW;
  u16b* Wvb = Wkb + NW;
  u16b* Wob = Wvb + NW;
  u16b* Qb  = Wob + NW;
  u16b* Kb  = Qb  + NX;
  u16b* Vb  = Kb  + NX;
  u16b* Ob  = Vb  + NX;   // total 48 MiB of workspace

  cvt_kernel<<<1024, 256, 0, stream>>>(x,  xb,  (int)NX);
  cvt_kernel<<<512,  256, 0, stream>>>(Wq, Wqb, (int)NW);
  cvt_kernel<<<512,  256, 0, stream>>>(Wk, Wkb, (int)NW);
  cvt_kernel<<<512,  256, 0, stream>>>(Wv, Wvb, (int)NW);
  cvt_kernel<<<512,  256, 0, stream>>>(Wo, Wob, (int)NW);

  dim3 ggrid(C_ / 128, M_ / 128);   // (8, 32)
  gemm_bf16<1><<<ggrid, 256, 0, stream>>>(xb, Wqb, bq, nullptr, Qb);
  gemm_bf16<1><<<ggrid, 256, 0, stream>>>(xb, Wkb, bk, nullptr, Kb);
  gemm_bf16<1><<<ggrid, 256, 0, stream>>>(xb, Wvb, bv, nullptr, Vb);

  rotary_kernel<<<(B_ * H_ * T_) / 256, 256, 0, stream>>>(Qb, Kb);

  dim3 agrid(T_ / 128, H_, B_);     // (16, 16, 2)
  attn_kernel<<<agrid, 256, 0, stream>>>(Qb, Kb, Vb, Ob);

  gemm_bf16<0><<<ggrid, 256, 0, stream>>>(Ob, Wob, bo, out, nullptr);
}
